// GNOCustomEncode_51075751084701
// MI455X (gfx1250) — matrix operand order repaired and checked by reading, not compiled, not run on hardware
//
#include <hip/hip_runtime.h>

#define NN 100000
#define NE 1600000

typedef __attribute__((ext_vector_type(16))) __bf16 v16bf;
typedef __attribute__((ext_vector_type(8)))  float  v8f;

union FragU { uint4 q[2]; v16bf v; };

__device__ __forceinline__ unsigned short f2bf(float f) {
    union { float f; unsigned u; } c;
    c.f = f;
    unsigned u = c.u;
    u += 0x7fffu + ((u >> 16) & 1u);   // round-to-nearest-even
    return (unsigned short)(u >> 16);
}

__global__ void k_zero(float* __restrict__ p, long long n) {
    long long i = (long long)blockIdx.x * blockDim.x + threadIdx.x;
    if (i < n) p[i] = 0.f;
}

// Repack a [K,128] fp32 weight into WMMA bf16 B-fragment layout:
// out[((kt*8+nt)*32+lane)*8+v] packs K=(kt*32 + (lane<16?0:8) + 2v + (v<4?0:8), +1) of column nt*16+(lane&15).
// mode 0: Wd = W1[0:64]-W1[64:128]; mode 1: Wb = W1[64:128]; mode 2: W3.
__global__ void k_repack(const float* __restrict__ W, int KT, int mode, unsigned* __restrict__ out) {
    int total = KT * 2048;
    int i = blockIdx.x * blockDim.x + threadIdx.x;
    if (i >= total) return;
    int v    = i & 7;
    int lane = (i >> 3) & 31;
    int nt   = (i >> 8) & 7;
    int kt   = i >> 11;
    int n  = nt * 16 + (lane & 15);
    int k0 = kt * 32 + ((lane & 16) ? 8 : 0) + 2 * v + ((v & 4) ? 8 : 0);
    float w0, w1;
    if (mode == 0) {
        w0 = W[k0 * 128 + n] - W[(k0 + 64) * 128 + n];
        w1 = W[(k0 + 1) * 128 + n] - W[(k0 + 65) * 128 + n];
    } else if (mode == 1) {
        w0 = W[(k0 + 64) * 128 + n];
        w1 = W[(k0 + 65) * 128 + n];
    } else {
        w0 = W[k0 * 128 + n];
        w1 = W[(k0 + 1) * 128 + n];
    }
    out[i] = (unsigned)f2bf(w0) | ((unsigned)f2bf(w1) << 16);
}

// Edge pass 1: s[dst] += x[src]; cnt[dst] += 1.  16 threads/edge, float4 per thread.
__global__ void k_edge1(const int* __restrict__ ei, const float* __restrict__ x,
                        float* __restrict__ s, float* __restrict__ cnt) {
    long long tid = (long long)blockIdx.x * blockDim.x + threadIdx.x;
    long long e = tid >> 4;
    if (e >= NE) return;
    int g   = (int)(tid & 15);
    int src = ei[e];
    int dst = ei[NE + e];
    float4 v = ((const float4*)(x + (long long)src * 64))[g];
    float* sp = s + (long long)dst * 64 + g * 4;
    atomicAdd(sp + 0, v.x);
    atomicAdd(sp + 1, v.y);
    atomicAdd(sp + 2, v.z);
    atomicAdd(sp + 3, v.w);
    if (g == 0) atomicAdd(cnt + dst, 1.f);
}

// Node pass: h2 = gelu(mask*(x@Wd + (s/c)@Wb + b1)) @ W3 + b3, all via v_wmma_f32_16x16x32_bf16.
// One wave per 16-node tile (N = 6250 tiles exactly).
__global__ __launch_bounds__(32) void k_node(
    const float* __restrict__ x, const float* __restrict__ s, const float* __restrict__ cnt,
    const unsigned* __restrict__ Wd, const unsigned* __restrict__ Wb, const unsigned* __restrict__ W3p,
    const float* __restrict__ b1, const float* __restrict__ b3, float* __restrict__ h2)
{
    __shared__ unsigned short Xi[16 * 64];   // bf16 x tile
    __shared__ unsigned short Sm[16 * 64];   // bf16 mean-neighbor tile
    __shared__ unsigned short G [16 * 128];  // bf16 gelu(mid)
    __shared__ float rvalid[16];

    const int lane = threadIdx.x;
    const long long base = (long long)blockIdx.x * 16;

    if (lane < 16) rvalid[lane] = (cnt[base + lane] > 0.f) ? 1.f : 0.f;
    for (int idx = lane; idx < 16 * 64; idx += 32) {
        int r = idx >> 6;
        long long node = base + r;
        float c = cnt[node];
        float inv = (c > 0.f) ? (1.f / c) : 0.f;
        Xi[idx] = f2bf(x[node * 64 + (idx & 63)]);
        Sm[idx] = f2bf(s[node * 64 + (idx & 63)] * inv);
    }
    __syncthreads();

    // A-fragment layout (16-bit, 16x32): lane<16 -> M=lane, K chunks {0..7, 16..23};
    // lane>=16 -> M=lane-16, K chunks {8..15, 24..31}.
    const int mrow  = lane & 15;
    const int base0 = (lane & 16) ? 8 : 0;
    const int hi8   = (lane & 16) ? 8 : 0;

    // GEMM1: mid[16,128] = Xi@Wd + Sm@Wb (+b1), K=64 -> 2 k-tiles of 32
    for (int nt = 0; nt < 8; ++nt) {
        v8f acc = {0.f, 0.f, 0.f, 0.f, 0.f, 0.f, 0.f, 0.f};
        for (int kt = 0; kt < 2; ++kt) {
            FragU a, b;
            const unsigned short* ap = Xi + mrow * 64 + kt * 32 + base0;
            a.q[0] = *(const uint4*)(ap);
            a.q[1] = *(const uint4*)(ap + 16);
            const uint4* wp = (const uint4*)(Wd + (((kt * 8 + nt) * 32 + lane) << 3));
            b.q[0] = wp[0]; b.q[1] = wp[1];
            acc = __builtin_amdgcn_wmma_f32_16x16x32_bf16(false, a.v, false, b.v,
                                                          (short)0, acc, false, false);
            const unsigned short* sp = Sm + mrow * 64 + kt * 32 + base0;
            a.q[0] = *(const uint4*)(sp);
            a.q[1] = *(const uint4*)(sp + 16);
            const uint4* wq = (const uint4*)(Wb + (((kt * 8 + nt) * 32 + lane) << 3));
            b.q[0] = wq[0]; b.q[1] = wq[1];
            acc = __builtin_amdgcn_wmma_f32_16x16x32_bf16(false, a.v, false, b.v,
                                                          (short)0, acc, false, false);
        }
        // C layout: lane<16 -> M=v, N=lane; lane>=16 -> M=v+8, N=lane-16
        int ncol = nt * 16 + (lane & 15);
        float bias = b1[ncol];
        for (int vv = 0; vv < 8; ++vv) {
            int m = vv + hi8;
            float t = (acc[vv] + bias) * rvalid[m];        // isolated nodes -> 0
            t = 0.5f * t * (1.f + erff(t * 0.70710678118654752f));  // exact-erf gelu
            G[m * 128 + ncol] = f2bf(t);
        }
    }
    __syncthreads();

    // GEMM2: h2[16,128] = G @ W3 (+b3), K=128 -> 4 k-tiles
    for (int nt = 0; nt < 8; ++nt) {
        v8f acc = {0.f, 0.f, 0.f, 0.f, 0.f, 0.f, 0.f, 0.f};
        for (int kt = 0; kt < 4; ++kt) {
            FragU a, b;
            const unsigned short* gp = G + mrow * 128 + kt * 32 + base0;
            a.q[0] = *(const uint4*)(gp);
            a.q[1] = *(const uint4*)(gp + 16);
            const uint4* wp = (const uint4*)(W3p + (((kt * 8 + nt) * 32 + lane) << 3));
            b.q[0] = wp[0]; b.q[1] = wp[1];
            acc = __builtin_amdgcn_wmma_f32_16x16x32_bf16(false, a.v, false, b.v,
                                                          (short)0, acc, false, false);
        }
        int ncol = nt * 16 + (lane & 15);
        float bias = b3[ncol];
        for (int vv = 0; vv < 8; ++vv) {
            int m = vv + hi8;
            h2[(base + m) * 128 + ncol] = acc[vv] + bias;
        }
    }
}

// Edge pass 2: out[dst] += h2[src].  32 threads/edge, float4 each.
__global__ void k_edge2(const int* __restrict__ ei, const float* __restrict__ h2,
                        float* __restrict__ out) {
    long long tid = (long long)blockIdx.x * blockDim.x + threadIdx.x;
    long long e = tid >> 5;
    if (e >= NE) return;
    int g   = (int)(tid & 31);
    int src = ei[e];
    int dst = ei[NE + e];
    float4 v = ((const float4*)(h2 + (long long)src * 128))[g];
    float* op = out + (long long)dst * 128 + g * 4;
    atomicAdd(op + 0, v.x);
    atomicAdd(op + 1, v.y);
    atomicAdd(op + 2, v.z);
    atomicAdd(op + 3, v.w);
}

__global__ void k_final(float* __restrict__ out, const float* __restrict__ cnt) {
    long long i = (long long)blockIdx.x * blockDim.x + threadIdx.x;
    if (i >= (long long)NN * 128) return;
    float c = cnt[i >> 7];
    out[i] *= 1.f / fmaxf(c, 1.f);
}

extern "C" void kernel_launch(void* const* d_in, const int* in_sizes, int n_in,
                              void* d_out, int out_size, void* d_ws, size_t ws_size,
                              hipStream_t stream) {
    (void)in_sizes; (void)n_in; (void)out_size; (void)ws_size;
    const float* x  = (const float*)d_in[0];
    const int*   ei = (const int*)d_in[1];
    const float* W1 = (const float*)d_in[2];
    const float* b1 = (const float*)d_in[3];
    const float* W3 = (const float*)d_in[4];
    const float* b3 = (const float*)d_in[5];
    float* out = (float*)d_out;

    char* ws = (char*)d_ws;
    float* s   = (float*)ws;  ws += (size_t)NN * 64 * sizeof(float);
    float* cnt = (float*)ws;  ws += (size_t)NN * sizeof(float);
    float* h2  = (float*)ws;  ws += (size_t)NN * 128 * sizeof(float);
    unsigned* Wd  = (unsigned*)ws;  ws += (size_t)4096 * sizeof(unsigned);
    unsigned* Wb  = (unsigned*)ws;  ws += (size_t)4096 * sizeof(unsigned);
    unsigned* W3p = (unsigned*)ws;

    const int T = 256;
    // zero s+cnt (adjacent) and out
    {
        long long n1 = (long long)NN * 65;
        k_zero<<<(unsigned)((n1 + T - 1) / T), T, 0, stream>>>(s, n1);
        long long n2 = (long long)NN * 128;
        k_zero<<<(unsigned)((n2 + T - 1) / T), T, 0, stream>>>(out, n2);
    }
    // weight repack (bf16 B-fragment layout)
    k_repack<<<(4096 + T - 1) / T, T, 0, stream>>>(W1, 2, 0, Wd);
    k_repack<<<(4096 + T - 1) / T, T, 0, stream>>>(W1, 2, 1, Wb);
    k_repack<<<(8192 + T - 1) / T, T, 0, stream>>>(W3, 4, 2, W3p);
    // edge pass 1: neighbor sum + degrees
    {
        long long th = (long long)NE * 16;
        k_edge1<<<(unsigned)((th + T - 1) / T), T, 0, stream>>>(ei, x, s, cnt);
    }
    // node pass: dual WMMA GEMM chain, one wave per 16-node tile
    k_node<<<NN / 16, 32, 0, stream>>>(x, s, cnt, Wd, Wb, W3p, b1, b3, h2);
    // edge pass 2: scatter h2[src] to out
    {
        long long th = (long long)NE * 32;
        k_edge2<<<(unsigned)((th + T - 1) / T), T, 0, stream>>>(ei, h2, out);
    }
    // mean normalize
    {
        long long th = (long long)NN * 128;
        k_final<<<(unsigned)((th + T - 1) / T), T, 0, stream>>>(out, cnt);
    }
}
